// DiffUnpool_3504693314190
// MI455X (gfx1250) — matrix operand order repaired and checked by reading, not compiled, not run on hardware
//
#include <hip/hip_runtime.h>

// ---- CDNA5 WMMA vector types (wave32) ----
typedef __attribute__((ext_vector_type(16))) __bf16 v16bf;
typedef __attribute__((ext_vector_type(8)))  float  v8f;
typedef int v4i __attribute__((vector_size(16)));   // matches builtin prototype

#define AS1 __attribute__((address_space(1)))
#define AS3 __attribute__((address_space(3)))

// Problem sizes (fixed by the reference): out[b] = S[b] @ x[b]
#define B_  16
#define M_  2048   // N_orig
#define K_  256    // N_pool
#define N_  256    // C

// Tiling
#define BM   128
#define BN   64
#define BK   32    // one v_wmma_f32_16x16x32_bf16 K-step
#define PK   36    // padded row pitch (floats): 36*4B=144B -> conflict-free b128 lane reads

// per-wave async ops issued per chunk: 4x b128 (S tile) + 8x b32 (x transpose) = 12
#define ASYNC_PER_CHUNK 12

#if __has_builtin(__builtin_amdgcn_global_load_async_to_lds_b128) && \
    __has_builtin(__builtin_amdgcn_global_load_async_to_lds_b32)
#define HAVE_ASYNC_LDS 1
#else
#define HAVE_ASYNC_LDS 0
#endif

__device__ __forceinline__ void cp_b128(const float* g, float* l) {
#if HAVE_ASYNC_LDS
    __builtin_amdgcn_global_load_async_to_lds_b128((AS1 v4i*)g, (AS3 v4i*)l, 0, 0);
#else
    *(float4*)l = *(const float4*)g;
#endif
}

__device__ __forceinline__ void cp_b32(const float* g, float* l) {
#if HAVE_ASYNC_LDS
    __builtin_amdgcn_global_load_async_to_lds_b32((AS1 int*)g, (AS3 int*)l, 0, 0);
#else
    *l = *g;
#endif
}

// wait ASYNCcnt <= ASYNC_PER_CHUNK: previous chunk's copies retired (in-order),
// next chunk's copies may still be in flight (overlap with WMMA compute)
__device__ __forceinline__ void async_wait_prev() {
#if HAVE_ASYNC_LDS
#if __has_builtin(__builtin_amdgcn_s_wait_asynccnt)
    __builtin_amdgcn_s_wait_asynccnt(ASYNC_PER_CHUNK);
#else
    asm volatile("s_wait_asynccnt 12" ::: "memory");
#endif
#endif
}

__device__ __forceinline__ void async_wait_zero() {
#if HAVE_ASYNC_LDS
#if __has_builtin(__builtin_amdgcn_s_wait_asynccnt)
    __builtin_amdgcn_s_wait_asynccnt(0);
#else
    asm volatile("s_wait_asynccnt 0" ::: "memory");
#endif
#endif
}

// Issue one chunk's async copies: S tile (contiguous b128, coalesced) and
// x tile (b32 scatter: consecutive lanes read consecutive n -> coalesced 128B
// global requests; per-lane LDS address performs the [k][n]->[n][k] transpose)
__device__ __forceinline__ void stage_tiles(const float* Sb, const float* xb,
                                            int m0, int n0, int k0, int t,
                                            float (*Ss)[PK], float (*Xs)[PK]) {
    #pragma unroll
    for (int p = 0; p < 4; ++p) {
        const int i  = t + p * 256;    // 0..1023 b128 chunks
        const int r  = i >> 3;         // row 0..127 (8 chunks per 32-float row)
        const int s4 = (i & 7) * 4;    // float offset 0..28
        cp_b128(Sb + (size_t)(m0 + r) * K_ + k0 + s4, &Ss[r][s4]);
    }
    const int n  = t & 63;
    const int kb = t >> 6;             // 0..3
    #pragma unroll
    for (int p = 0; p < 8; ++p) {
        const int k = kb + p * 4;      // 0..31
        cp_b32(xb + (size_t)(k0 + k) * N_ + n0 + n, &Xs[n][k]);
    }
}

// fp32x4 quads -> one 16-element bf16 fragment (lowers to v_cvt_pk_bf16_f32 pairs)
__device__ __forceinline__ v16bf cvt16(float4 a, float4 b, float4 c, float4 d) {
    v16bf r;
    r[0]  = (__bf16)a.x; r[1]  = (__bf16)a.y; r[2]  = (__bf16)a.z; r[3]  = (__bf16)a.w;
    r[4]  = (__bf16)b.x; r[5]  = (__bf16)b.y; r[6]  = (__bf16)b.z; r[7]  = (__bf16)b.w;
    r[8]  = (__bf16)c.x; r[9]  = (__bf16)c.y; r[10] = (__bf16)c.z; r[11] = (__bf16)c.w;
    r[12] = (__bf16)d.x; r[13] = (__bf16)d.y; r[14] = (__bf16)d.z; r[15] = (__bf16)d.w;
    return r;
}

__global__ __launch_bounds__(256) void diffunpool_bf16_wmma(
    const float* __restrict__ x,   // (B, K_, N_)
    const float* __restrict__ S,   // (B, M_, K_)
    float* __restrict__ out)       // (B, M_, N_)
{
    // Double-buffered fp32 staging (async engine moves raw bytes); bf16 at fragment build.
    __shared__ __align__(16) float Ssf[2][BM][PK];  // 2 x 18 KB
    __shared__ __align__(16) float Xsf[2][BN][PK];  // 2 x  9 KB

    const int t  = threadIdx.x;
    const int b  = blockIdx.z;
    const int m0 = blockIdx.x * BM;
    const int n0 = blockIdx.y * BN;

    const int lane  = t & 31;
    const int wid   = t >> 5;          // 8 waves
    const int wm    = wid & 3;         // wave M index (0..3)
    const int wn    = wid >> 2;        // wave N index (0..1)
    const int lmod  = lane & 15;
    const int lhalf = lane >> 4;       // 0: lanes 0-15, 1: lanes 16-31

    const float* Sb = S   + (size_t)b * M_ * K_;
    const float* xb = x   + (size_t)b * K_ * N_;
    float*       ob = out + (size_t)b * M_ * N_;

    v8f c00 = {}, c01 = {}, c10 = {}, c11 = {};

    // prologue: start chunk 0 copies
    stage_tiles(Sb, xb, m0, n0, 0, t, Ssf[0], Xsf[0]);

    for (int k0 = 0; k0 < K_; k0 += BK) {
        const int cur = (k0 >> 5) & 1;

        if (k0 + BK < K_) {
            // kick off chunk k0+BK into the other buffer, then wait only for
            // chunk k0's 12 per-wave async ops (they retire in order)
            stage_tiles(Sb, xb, m0, n0, k0 + BK, t, Ssf[cur ^ 1], Xsf[cur ^ 1]);
            async_wait_prev();
        } else {
            async_wait_zero();
        }
        __syncthreads();    // all waves' chunk-k0 LDS writes visible

        // ---- fragments per CDNA5 WMMA VGPR layouts (bf16 cvt here) ----
        // A (16x32): lane<16 holds K {0..7,16..23}, lane>=16 holds {8..15,24..31}
        const float (*Ss)[PK] = Ssf[cur];
        const float (*Xs)[PK] = Xsf[cur];
        const int mA0 = wm * 32 + lmod;
        const int mA1 = mA0 + 16;
        const int ka  = lhalf * 8;
        const v16bf a0 = cvt16(*(const float4*)&Ss[mA0][ka],      *(const float4*)&Ss[mA0][ka + 4],
                               *(const float4*)&Ss[mA0][16 + ka], *(const float4*)&Ss[mA0][20 + ka]);
        const v16bf a1 = cvt16(*(const float4*)&Ss[mA1][ka],      *(const float4*)&Ss[mA1][ka + 4],
                               *(const float4*)&Ss[mA1][16 + ka], *(const float4*)&Ss[mA1][20 + ka]);
        // B (32x16): lane<16 holds K {0..7,16..23} of column n, lane>=16 holds {8..15,24..31}
        const int nB0 = wn * 32 + lmod;
        const int nB1 = nB0 + 16;
        const int kb2 = lhalf * 8;
        const v16bf b0 = cvt16(*(const float4*)&Xs[nB0][kb2],     *(const float4*)&Xs[nB0][kb2 + 4],
                               *(const float4*)&Xs[nB0][kb2 + 16], *(const float4*)&Xs[nB0][kb2 + 20]);
        const v16bf b1 = cvt16(*(const float4*)&Xs[nB1][kb2],     *(const float4*)&Xs[nB1][kb2 + 4],
                               *(const float4*)&Xs[nB1][kb2 + 16], *(const float4*)&Xs[nB1][kb2 + 20]);

        // ---- 4 WMMAs: 32x32 output tile per wave, fp32 accumulation ----
        // (overlaps with the in-flight async copies of the next chunk)
        c00 = __builtin_amdgcn_wmma_f32_16x16x32_bf16(false, a0, false, b0, (short)0, c00, false, false);
        c01 = __builtin_amdgcn_wmma_f32_16x16x32_bf16(false, a0, false, b1, (short)0, c01, false, false);
        c10 = __builtin_amdgcn_wmma_f32_16x16x32_bf16(false, a1, false, b0, (short)0, c10, false, false);
        c11 = __builtin_amdgcn_wmma_f32_16x16x32_bf16(false, a1, false, b1, (short)0, c11, false, false);

        __syncthreads();    // next iteration's staging overwrites buffer cur
    }

    // ---- epilogue: C/D layout is VGPR r -> M=r (lanes 0-15) / M=r+8 (lanes 16-31)
    const int mw = m0 + wm * 32;
    const int nw = n0 + wn * 32;
    #pragma unroll
    for (int r = 0; r < 8; ++r) {
        const int row = mw + r + lhalf * 8;
        const int col = nw + lmod;
        ob[(size_t)row        * N_ + col     ] = c00[r];
        ob[(size_t)row        * N_ + col + 16] = c01[r];
        ob[(size_t)(row + 16) * N_ + col     ] = c10[r];
        ob[(size_t)(row + 16) * N_ + col + 16] = c11[r];
    }
}

extern "C" void kernel_launch(void* const* d_in, const int* in_sizes, int n_in,
                              void* d_out, int out_size, void* d_ws, size_t ws_size,
                              hipStream_t stream) {
    (void)in_sizes; (void)n_in; (void)out_size; (void)d_ws; (void)ws_size;
    const float* x = (const float*)d_in[0];   // (16, 256, 256)
    const float* S = (const float*)d_in[1];   // (16, 2048, 256)
    // d_in[2] (A) is unused by the reference — never touched (saves 268 MB of traffic).
    float* out = (float*)d_out;               // (16, 2048, 256)

    dim3 grid(M_ / BM, N_ / BN, B_);          // (16, 4, 16)
    diffunpool_bf16_wmma<<<grid, 256, 0, stream>>>(x, S, out);
}
